// RelativeGeometryEncoding_21131239097221
// MI455X (gfx1250) — matrix operand order repaired and checked by reading, not compiled, not run on hardware
//
#include <hip/hip_runtime.h>
#include <hip/hip_bf16.h>
#include <math.h>

typedef __attribute__((ext_vector_type(16))) _Float16 v16h;
typedef __attribute__((ext_vector_type(8)))  _Float16 v8h;
typedef __attribute__((ext_vector_type(8)))  float    v8f;

#define NB      16
#define OUTD    128
#define ROWH    40   // halves per LDS feature row (80B pitch -> conflict-free A-tile b128 loads)

__global__ __launch_bounds__(256) void rge_wmma_kernel(
    const float* __restrict__ frame_t,
    const float* __restrict__ frame_R,
    const int*   __restrict__ edge_src,
    const int*   __restrict__ edge_dst,
    const float* __restrict__ Wm,      // [128, 31] row-major
    float* __restrict__ out,           // [E, 128]
    int E)
{
  // 8 waves/block, each wave owns a 32-row x 40-half slab (2560B) -> 20KB/block
  __shared__ __align__(16) _Float16 smem[8 * 32 * ROWH];

  const int lane = threadIdx.x & 31;
  const int wib  = threadIdx.x >> 5;
  _Float16* feat = smem + wib * 32 * ROWH;

  const int n  = lane & 15;   // WMMA column within tile / A-row selector
  const int kh = lane >> 4;   // K half-group (0 or 1)

  // ---- B operand: W^T in WMMA 32x16 f16 B layout, 8 tiles of 16 output cols ----
  // Unconditional loads (clamped index) + cndmask select -> no exec-mask churn.
  v16h B[8];
  #pragma unroll
  for (int t = 0; t < 8; ++t) {
    const float* wr = Wm + (t * 16 + n) * 31;
    #pragma unroll
    for (int j = 0; j < 16; ++j) {
      const int k  = kh * 16 + j;
      const int kc = (k < 31) ? k : 30;       // in-bounds dummy for the K=31 pad
      const float wv = wr[kc];
      B[t][j] = (_Float16)((k < 31) ? wv : 0.0f);
    }
  }

  // Scalarize wave id -> all chunk/loop/tile-bound math runs on SALU, and the
  // per-tile range branch is a scalar branch (WMMA always under full EXEC).
  const int wavesTotal = (int)((gridDim.x * blockDim.x) >> 5);
  const int waveId     = __builtin_amdgcn_readfirstlane(
                           (int)((blockIdx.x * blockDim.x + threadIdx.x) >> 5));
  const int nChunks    = (E + 31) >> 5;

  for (int chunk = waveId; chunk < nChunks; chunk += wavesTotal) {
    const int e  = chunk * 32 + lane;
    const int ec = (e < E) ? e : (E - 1);   // clamp; OOB rows never stored

    // prefetch next chunk's edge indices (global_prefetch_b8)
    {
      const int nc = chunk + wavesTotal;
      if (nc < nChunks) {
        __builtin_prefetch(edge_src + nc * 32 + lane, 0, 0);
        __builtin_prefetch(edge_dst + nc * 32 + lane, 0, 0);
      }
    }

    // ---- per-lane feature computation (f32 math, pack to f16), unconditional ----
    union { _Float16 h[32]; v8h v[4]; } F;
    {
      const int s = edge_src[ec];
      const int d = edge_dst[ec];
      const float* ts = frame_t + s * 3;
      const float* td = frame_t + d * 3;
      const float* rs = frame_R + s * 9;
      const float* rd = frame_R + d * 9;
      float Rs[9], Rd[9];
      #pragma unroll
      for (int i = 0; i < 9; ++i) Rs[i] = rs[i];
      #pragma unroll
      for (int i = 0; i < 9; ++i) Rd[i] = rd[i];

      const float dx = ts[0] - td[0];
      const float dy = ts[1] - td[1];
      const float dz = ts[2] - td[2];
      const float d2   = dx*dx + dy*dy + dz*dz;
      const float dist = sqrtf(d2);
      const float invs = rsqrtf(d2 + 1.0f);

      // K 0..15: Gaussian RBF, centers linspace(0,20,16) -> spacing 20/15
      #pragma unroll
      for (int b = 0; b < NB; ++b) {
        const float u = dist - (20.0f / 15.0f) * (float)b;
        F.h[b] = (_Float16)__expf(-u * u);
      }
      // K 16..18: dir_l = (-diff)^T Rs * invs
      #pragma unroll
      for (int j = 0; j < 3; ++j)
        F.h[16 + j] = (_Float16)(-(dx*Rs[j] + dy*Rs[3+j] + dz*Rs[6+j]) * invs);
      // K 19..21: dir_r = diff^T Rd * invs
      #pragma unroll
      for (int j = 0; j < 3; ++j)
        F.h[19 + j] = (_Float16)((dx*Rd[j] + dy*Rd[3+j] + dz*Rd[6+j]) * invs);
      // K 22..30: orient[i][j] = col_i(Rs) . col_j(Rd)
      #pragma unroll
      for (int i = 0; i < 3; ++i)
        #pragma unroll
        for (int j = 0; j < 3; ++j)
          F.h[22 + i*3 + j] =
            (_Float16)(Rs[i]*Rd[j] + Rs[3+i]*Rd[3+j] + Rs[6+i]*Rd[6+j]);
      F.h[31] = (_Float16)0.0f;  // K pad
    }

    // write feature row to this wave's LDS slab (4x ds_store_b128)
    {
      _Float16* row = feat + lane * ROWH;
      #pragma unroll
      for (int q = 0; q < 4; ++q)
        *(v8h*)(row + q * 8) = F.v[q];
    }

    // same-wave LDS is in-order; fence the compiler and drain DScnt
    __builtin_amdgcn_wave_barrier();
    asm volatile("s_wait_dscnt 0" ::: "memory");
    __builtin_amdgcn_wave_barrier();

    // ---- two 16-edge A tiles -> 8 WMMAs each ----
    #pragma unroll
    for (int at = 0; at < 2; ++at) {
      // A 16x32 f16, contiguous k order used here: lanes 0-15 K=0..15; lanes 16-31 K=16..31 (matches B)
      const _Float16* ar = feat + (at * 16 + n) * ROWH + kh * 16;
      const v8h lo = *(const v8h*)(ar);        // K = kh*16 .. kh*16+7
      const v8h hi = *(const v8h*)(ar + 8);    // K = kh*16+8 ..
      v16h a;
      #pragma unroll
      for (int j = 0; j < 8; ++j) { a[j] = lo[j]; a[8 + j] = hi[j]; }

      const int ebase = chunk * 32 + at * 16;  // SGPR (chunk is scalar)
      const int moff  = kh * 8;                // C/D layout: lanes>=16 hold rows m+8

      // one per-lane base pointer; all 64 stores become base + const byte offsets
      float* basep = out + (long)(ebase + moff) * OUTD + n;

      if (ebase + 16 <= E) {
        // hot path (always taken when E % 32 == 0): scalar branch, no per-store guards
        #pragma unroll
        for (int t = 0; t < 8; ++t) {
          v8f acc;
          #pragma unroll
          for (int j = 0; j < 8; ++j) acc[j] = 0.0f;
          acc = __builtin_amdgcn_wmma_f32_16x16x32_f16(
              false, a, false, B[t], (short)0, acc, false, false);
          #pragma unroll
          for (int j = 0; j < 8; ++j)
            __builtin_nontemporal_store(acc[j], basep + j * OUTD + t * 16);
        }
      } else if (ebase < E) {
        // tail tile: per-row guard (cold)
        #pragma unroll
        for (int t = 0; t < 8; ++t) {
          v8f acc;
          #pragma unroll
          for (int j = 0; j < 8; ++j) acc[j] = 0.0f;
          acc = __builtin_amdgcn_wmma_f32_16x16x32_f16(
              false, a, false, B[t], (short)0, acc, false, false);
          #pragma unroll
          for (int j = 0; j < 8; ++j)
            if (ebase + moff + j < E)
              __builtin_nontemporal_store(acc[j], basep + j * OUTD + t * 16);
        }
      }
    }
  }
}

extern "C" void kernel_launch(void* const* d_in, const int* in_sizes, int n_in,
                              void* d_out, int out_size, void* d_ws, size_t ws_size,
                              hipStream_t stream) {
  const float* frame_t  = (const float*)d_in[0];
  const float* frame_R  = (const float*)d_in[1];
  const int*   edge_src = (const int*)d_in[2];
  const int*   edge_dst = (const int*)d_in[3];
  const float* Wm       = (const float*)d_in[4];
  float*       out      = (float*)d_out;
  const int E = in_sizes[2];

  const int threads = 256;                 // 8 wave32
  const int nChunks = (E + 31) / 32;
  int blocks = (nChunks + 7) / 8;
  if (blocks > 1024) blocks = 1024;        // grid-stride: ~6 chunks/wave at E=1.6M,
  if (blocks < 1)    blocks = 1;           // amortizes the W->B register build
  rge_wmma_kernel<<<blocks, threads, 0, stream>>>(
      frame_t, frame_R, edge_src, edge_dst, Wm, out, E);
}
